// RotarySelfAttention_9740985827952
// MI455X (gfx1250) — compile-verified, matrix operand order checked by reading, not run on hardware
//
#include <hip/hip_runtime.h>
#include <hip/hip_bf16.h>
#include <stdint.h>

// ---------------- problem constants ----------------
#define NB   4
#define NSEQ 2048
#define DIM  1024
#define NH   16
#define DHD  64
#define INNER (NH*DHD)          // 1024
#define MROWS (NB*NSEQ)         // 8192
#define NEGBIG (-1.0e9f)

// ---------------- vector types ----------------
typedef __bf16 bf16x16 __attribute__((ext_vector_type(16)));
typedef float  f32x8   __attribute__((ext_vector_type(8)));
typedef unsigned short u16x8 __attribute__((ext_vector_type(8)));
typedef unsigned short u16x4 __attribute__((ext_vector_type(4)));
typedef unsigned char  u8x8  __attribute__((ext_vector_type(8)));

// ---------------- helpers ----------------
__device__ __forceinline__ unsigned short f2bfu(float f) {
  union { float f; unsigned int u; } x; x.f = f;
  unsigned int r = x.u + 0x7fffu + ((x.u >> 16) & 1u);   // RNE (NaN ignored)
  return (unsigned short)(r >> 16);
}
__device__ __forceinline__ __bf16 f2bf16(float f) {
  union { unsigned short u; __bf16 b; } c; c.u = f2bfu(f);
  return c.b;
}

// Load one 16x32 bf16 fragment (A layout; B layout assumed mirrored).
// lane<16: row=lane, K={0..7,16..23}; lane>=16: row=lane-16, K={8..15,24..31}.
// Works for global or LDS (generic pointer -> global_load_b128 / ds_load_b128).
__device__ __forceinline__ bf16x16 load_frag_k32(const unsigned short* base,
                                                 int ld, int lane) {
  int row = lane & 15;
  int kh  = (lane >> 4) << 3;
  const unsigned short* p = base + (size_t)row * ld + kh;
  union { bf16x16 v; u16x8 h[2]; } u;
  u.h[0] = *(const u16x8*)(p);
  u.h[1] = *(const u16x8*)(p + 16);
  return u.v;
}

__device__ __forceinline__ f32x8 wmma_bf16(bf16x16 a, bf16x16 b, f32x8 c) {
  return __builtin_amdgcn_wmma_f32_16x16x32_bf16(
      /*neg_a=*/false, a, /*neg_b=*/false, b,
      /*c_mod=*/(short)0, c, /*reuse_a=*/false, /*reuse_b=*/false);
}

// Async global -> LDS copy of 16 bytes (one b128 transfer), tracked by ASYNCcnt.
// Inline asm is the toolchain-portable route for the async/TDM family.
__device__ __forceinline__ void async_ld_b128(unsigned lds_off, const void* g) {
  unsigned long long ga = (unsigned long long)(size_t)g;
  asm volatile("global_load_async_to_lds_b128 %0, %1, off"
               :: "v"(lds_off), "v"(ga) : "memory");
}
__device__ __forceinline__ unsigned lds_addr_of(const void* p) {
  return (unsigned)(size_t)p;   // generic shared ptr: low 32 bits = LDS offset
}

// ---------------- 1) fp32 -> bf16 weight conversion ----------------
__global__ void k_convert_weights(const float* __restrict__ wqkv,
                                  const float* __restrict__ wout,
                                  unsigned short* __restrict__ wqkv_bf,
                                  unsigned short* __restrict__ wout_bf) {
  const int NQKV = 3 * INNER * DIM;            // 3145728
  int base = (blockIdx.x * blockDim.x + threadIdx.x) * 4;
  const float* src; unsigned short* dst; int off;
  if (base < NQKV) { src = wqkv; dst = wqkv_bf; off = base; }
  else             { src = wout; dst = wout_bf; off = base - NQKV; }
  float4 v = *(const float4*)(src + off);
  u16x4 o;
  o[0] = f2bfu(v.x); o[1] = f2bfu(v.y); o[2] = f2bfu(v.z); o[3] = f2bfu(v.w);
  *(u16x4*)(dst + off) = o;
}

// ---------------- 2) cos/sin tables ----------------
__global__ void k_trig(const float* __restrict__ emb, float* __restrict__ ct,
                       float* __restrict__ st, int n) {
  int i = blockIdx.x * blockDim.x + threadIdx.x;
  if (i < n) { float e = emb[i]; ct[i] = __cosf(e); st[i] = __sinf(e); }
}

// ---------------- 3) LayerNorm -> bf16 ----------------
__global__ void __launch_bounds__(256) k_layernorm(
    const float* __restrict__ x, const float* __restrict__ gamma,
    const float* __restrict__ beta, unsigned short* __restrict__ xn) {
  int row = blockIdx.x, tid = threadIdx.x;
  const float* xr = x + (size_t)row * DIM;
  float4 v = *(const float4*)(xr + tid * 4);
  float s = v.x + v.y + v.z + v.w;
  float q = v.x*v.x + v.y*v.y + v.z*v.z + v.w*v.w;
  __shared__ float ss[256], sq[256];
  ss[tid] = s; sq[tid] = q; __syncthreads();
  for (int o = 128; o > 0; o >>= 1) {
    if (tid < o) { ss[tid] += ss[tid + o]; sq[tid] += sq[tid + o]; }
    __syncthreads();
  }
  float mu   = ss[0] * (1.0f / DIM);
  float var  = sq[0] * (1.0f / DIM) - mu * mu;
  float rstd = rsqrtf(var + 1e-5f);
  float4 g = *(const float4*)(gamma + tid * 4);
  float4 b = *(const float4*)(beta  + tid * 4);
  u16x4 o;
  o[0] = f2bfu((v.x - mu) * rstd * g.x + b.x);
  o[1] = f2bfu((v.y - mu) * rstd * g.y + b.y);
  o[2] = f2bfu((v.z - mu) * rstd * g.z + b.z);
  o[3] = f2bfu((v.w - mu) * rstd * g.w + b.w);
  *(u16x4*)(xn + (size_t)row * DIM + tid * 4) = o;
}

// ---------------- 4/7) bf16 GEMM: C[m][n] = sum_k A[m][k]*B[n][k] (+bias) ----
// A: MxK row-major bf16, B: NxK row-major bf16, C: MxN fp32.
// One wave computes a 64x64 tile via 4x4 grid of 16x16x32 WMMAs.
__global__ void __launch_bounds__(256) k_gemm_bf16(
    const unsigned short* __restrict__ A, const unsigned short* __restrict__ B,
    float* __restrict__ C, const float* __restrict__ bias,
    int M, int N, int K) {
  int lane   = threadIdx.x & 31;
  int wid    = blockIdx.x * (blockDim.x >> 5) + (threadIdx.x >> 5);
  int tilesN = N >> 6;
  int tm = wid / tilesN, tn = wid % tilesN;
  if (tm * 64 >= M) return;
  const unsigned short* Abase = A + (size_t)tm * 64 * K;
  const unsigned short* Bbase = B + (size_t)tn * 64 * K;

  f32x8 acc[4][4] = {};
  for (int k0 = 0; k0 < K; k0 += 32) {
    if (k0 + 64 < K) {
      __builtin_prefetch(Abase + (size_t)k0 + 64, 0, 3);
      __builtin_prefetch(Bbase + (size_t)k0 + 64, 0, 3);
    }
    bf16x16 af[4], bg[4];
#pragma unroll
    for (int i = 0; i < 4; ++i)
      af[i] = load_frag_k32(Abase + (size_t)(i * 16) * K + k0, K, lane);
#pragma unroll
    for (int j = 0; j < 4; ++j)
      bg[j] = load_frag_k32(Bbase + (size_t)(j * 16) * K + k0, K, lane);
#pragma unroll
    for (int i = 0; i < 4; ++i)
#pragma unroll
      for (int j = 0; j < 4; ++j)
        acc[i][j] = wmma_bf16(af[i], bg[j], acc[i][j]);
  }
  int hi = lane >> 4, col = lane & 15;
#pragma unroll
  for (int i = 0; i < 4; ++i) {
#pragma unroll
    for (int j = 0; j < 4; ++j) {
      int n  = tn * 64 + j * 16 + col;
      int m0 = tm * 64 + i * 16 + 8 * hi;
      float bv = bias ? bias[n] : 0.0f;
      float* cp = C + (size_t)m0 * N + n;
#pragma unroll
      for (int e = 0; e < 8; ++e) cp[(size_t)e * N] = acc[i][j][e] + bv;
    }
  }
}

// ---------------- 5) RoPE + pack q,k [b][h][t][d], vT [b][h][d][t] ---------
__global__ void k_rope_pack(const float* __restrict__ qkv,
                            const float* __restrict__ ct,
                            const float* __restrict__ st,
                            unsigned short* __restrict__ qb,
                            unsigned short* __restrict__ kb,
                            unsigned short* __restrict__ vT) {
  long long p = (long long)blockIdx.x * blockDim.x + threadIdx.x; // pair index
  int m = (int)(p / 1536);          // row in [0,8192)
  int r = (int)(p % 1536);
  int e0 = r << 1;                  // even column in [0,3072)
  int which = e0 >> 10;             // 0=q,1=k,2=v
  int rem = e0 & 1023;
  int h = rem >> 6, d = rem & 63;   // d even
  int b = m >> 11, t = m & 2047;
  float x0 = qkv[(size_t)m * 3072 + e0];
  float x1 = qkv[(size_t)m * 3072 + e0 + 1];
  size_t ti = (size_t)m * 64 + d;
  float c0 = ct[ti], s0 = st[ti], c1 = ct[ti + 1], s1 = st[ti + 1];
  float y0 = x0 * c0 - x1 * s0;     // rope: even
  float y1 = x1 * c1 + x0 * s1;     // rope: odd
  unsigned int packed = (unsigned int)f2bfu(y0) | ((unsigned int)f2bfu(y1) << 16);
  int bh = b * 16 + h;
  if (which == 0)
    *(unsigned int*)(qb + ((size_t)bh * 2048 + t) * 64 + d) = packed;
  else if (which == 1)
    *(unsigned int*)(kb + ((size_t)bh * 2048 + t) * 64 + d) = packed;
  else {
    vT[((size_t)bh * 64 + d)     * 2048 + t] = f2bfu(y0);
    vT[((size_t)bh * 64 + d + 1) * 2048 + t] = f2bfu(y1);
  }
}

// ---------------- 6) transposed flash attention (LDS-staged K/V) ----------
// One block = 8 waves, all on the same (b,h); each wave owns 16 queries.
// K/V chunks (32 keys) are double-buffered in LDS via async b128 copies and
// shared by all 8 waves (8x traffic reduction). S^T = K*Q^T so P^T in
// C-layout is directly the B-fragment of Out^T = V^T * P^T (zero shuffles).
__global__ void __launch_bounds__(256) k_attention(
    const unsigned short* __restrict__ qb, const unsigned short* __restrict__ kb,
    const unsigned short* __restrict__ vT, const unsigned char* __restrict__ mask,
    const float* __restrict__ ct, const float* __restrict__ st,
    unsigned short* __restrict__ ao) {
  __shared__ unsigned short ks[2][32][64];   // 8 KB: keys x dims
  __shared__ unsigned short vs[2][64][32];   // 8 KB: dims x keys
  int tid  = threadIdx.x;
  int lane = tid & 31;
  int warp = tid >> 5;
  int bh    = blockIdx.x >> 4;               // 16 blocks per (b,h)
  int qtile = ((blockIdx.x & 15) << 3) + warp;
  int b = bh >> 4, h = bh & 15;
  int hi = lane >> 4, col = lane & 15;
  int t = qtile * 16 + col;                  // this lane's query index

  const unsigned short* qbase = qb + ((size_t)bh * 2048 + qtile * 16) * 64;
  const unsigned short* kbase = kb + (size_t)bh * 2048 * 64;
  const unsigned short* vbase = vT + (size_t)bh * 64 * 2048;
  const unsigned char*  mrow  = mask + (size_t)b * 2048;

  // per-thread async copy slots (one b128 each for K and V chunk)
  int kr = tid >> 3, kc = (tid & 7) << 3;    // K chunk: 32 rows x 64 dims
  int vd = tid >> 2, vc = (tid & 3) << 3;    // V chunk: 64 rows x 32 keys

  // Q as B-fragments (N = queries, K = head dim), direct global loads
  bf16x16 bq0 = load_frag_k32(qbase, 64, lane);       // d 0..31
  bf16x16 bq1 = load_frag_k32(qbase + 32, 64, lane);  // d 32..63

  // prologue: stage chunk 0 into buffer 0
  async_ld_b128(lds_addr_of(&ks[0][kr][kc]), kbase + (size_t)kr * 64 + kc);
  async_ld_b128(lds_addr_of(&vs[0][vd][vc]), vbase + (size_t)vd * 2048 + vc);

  f32x8 acc[4] = {};                 // Out^T: 4 d-tiles x (8 rows/lane)
  float m_run = -1e30f, l_run = 0.0f;

  for (int it = 0; it < 64; ++it) {
    int kt  = it << 5;
    int cur = it & 1;
    bool more = (it + 1) < 64;
    if (more) {
      int ktn = kt + 32;
      async_ld_b128(lds_addr_of(&ks[cur ^ 1][kr][kc]),
                    kbase + (size_t)(ktn + kr) * 64 + kc);
      async_ld_b128(lds_addr_of(&vs[cur ^ 1][vd][vc]),
                    vbase + (size_t)vd * 2048 + ktn + vc);
      // allow the 2 just-issued ops to stay outstanding; older ones complete
      asm volatile("s_wait_asynccnt 0x2" ::: "memory");
    } else {
      asm volatile("s_wait_asynccnt 0x0" ::: "memory");
    }
    __syncthreads();                 // current buffer visible to all waves

    const unsigned short* kcur = &ks[cur][0][0];
    const unsigned short* vcur = &vs[cur][0][0];

    // K rows as A-fragments (M = keys), read from LDS
    bf16x16 ka00 = load_frag_k32(kcur,            64, lane);
    bf16x16 ka01 = load_frag_k32(kcur + 32,       64, lane);
    bf16x16 ka10 = load_frag_k32(kcur + 16 * 64,      64, lane);
    bf16x16 ka11 = load_frag_k32(kcur + 16 * 64 + 32, 64, lane);
    f32x8 z = {};
    f32x8 s0 = wmma_bf16(ka00, bq0, wmma_bf16(ka01, bq1, z)); // keys kt..+15
    f32x8 s1 = wmma_bf16(ka10, bq0, wmma_bf16(ka11, bq1, z)); // keys kt+16..+31

    u8x8 mk0 = *(const u8x8*)(mrow + kt + 8 * hi);
    u8x8 mk1 = *(const u8x8*)(mrow + kt + 16 + 8 * hi);
    float p0[8], p1[8];
    float mx = -1e30f;
#pragma unroll
    for (int e = 0; e < 8; ++e) {
      float a0 = mk0[e] ? s0[e] * 0.125f : NEGBIG;
      float a1 = mk1[e] ? s1[e] * 0.125f : NEGBIG;
      p0[e] = a0; p1[e] = a1;
      mx = fmaxf(mx, fmaxf(a0, a1));
    }
    mx = fmaxf(mx, __shfl_xor(mx, 16));         // per-query max over 32 keys
    float m_new = fmaxf(m_run, mx);
    float corr  = __expf(m_run - m_new);
    float ls = 0.0f;
    bf16x16 bp;                                  // P^T as B-fragment
#pragma unroll
    for (int e = 0; e < 8; ++e) {
      float e0v = __expf(p0[e] - m_new);
      float e1v = __expf(p1[e] - m_new);
      ls += e0v + e1v;
      bp[e]     = f2bf16(e0v);
      bp[e + 8] = f2bf16(e1v);
    }
    ls += __shfl_xor(ls, 16);
    l_run = l_run * corr + ls;
    m_run = m_new;
#pragma unroll
    for (int j = 0; j < 4; ++j) {
      acc[j] *= corr;
      bf16x16 av = load_frag_k32(vcur + j * 16 * 32, 32, lane);
      acc[j] = wmma_bf16(av, bp, acc[j]);
    }
    __syncthreads();                 // all waves done reading before overwrite
  }

  // epilogue: 1/l, inverse RoPE (emb -> -emb), bf16 pack, one b128 store/tile
  float rinv = 1.0f / l_run;
  size_t bt = ((size_t)b * 2048 + t) * 64;
  unsigned short* aoro = ao + ((size_t)b * 2048 + t) * 1024 + h * 64;
#pragma unroll
  for (int j = 0; j < 4; ++j) {
    int dbase = j * 16 + 8 * hi;
    u16x8 outv;
#pragma unroll
    for (int e = 0; e < 8; e += 2) {
      int d0 = dbase + e;
      float x0 = acc[j][e] * rinv, x1 = acc[j][e + 1] * rinv;
      float c0 = ct[bt + d0],     ss0 = st[bt + d0];
      float c1 = ct[bt + d0 + 1], ss1 = st[bt + d0 + 1];
      outv[e]     = f2bfu(x0 * c0 + x1 * ss0);
      outv[e + 1] = f2bfu(x1 * c1 - x0 * ss1);
    }
    *(u16x8*)(aoro + dbase) = outv;
  }
}

// ---------------- launcher ----------------
extern "C" void kernel_launch(void* const* d_in, const int* in_sizes, int n_in,
                              void* d_out, int out_size, void* d_ws, size_t ws_size,
                              hipStream_t stream) {
  (void)in_sizes; (void)n_in; (void)out_size; (void)ws_size;
  const float* x     = (const float*)d_in[0];
  const float* emb   = (const float*)d_in[1];
  const unsigned char* xmask = (const unsigned char*)d_in[2];
  const float* gamma = (const float*)d_in[3];
  const float* beta  = (const float*)d_in[4];
  const float* wqkv  = (const float*)d_in[5];
  const float* wout  = (const float*)d_in[6];
  const float* bout  = (const float*)d_in[7];
  float* out = (float*)d_out;

  char* ws = (char*)d_ws;
  unsigned short* wqkv_bf = (unsigned short*)(ws + 0);          //  6 MB
  unsigned short* wout_bf = (unsigned short*)(ws + 6291456);    //  2 MB
  float*          ct      = (float*)(ws + 8388608);             //  2 MB
  float*          st      = (float*)(ws + 10485760);            //  2 MB
  unsigned short* xn      = (unsigned short*)(ws + 12582912);   // 16 MB
  float*          qkv     = (float*)(ws + 29360128);            // 96 MB
  unsigned short* qb      = (unsigned short*)(ws + 130023424);  // 16 MB
  unsigned short* kb      = (unsigned short*)(ws + 146800640);  // 16 MB
  unsigned short* vT      = (unsigned short*)(ws + 163577856);  // 16 MB
  unsigned short* ao      = (unsigned short*)(ws + 180355072);  // 16 MB

  k_convert_weights<<<4096, 256, 0, stream>>>(wqkv, wout, wqkv_bf, wout_bf);
  k_trig<<<2048, 256, 0, stream>>>(emb, ct, st, NB * NSEQ * DHD);
  k_layernorm<<<MROWS, 256, 0, stream>>>(x, gamma, beta, xn);
  // qkv = xn @ wqkv^T : M=8192 N=3072 K=1024 -> 128*48 = 6144 waves
  k_gemm_bf16<<<768, 256, 0, stream>>>(xn, wqkv_bf, qkv, nullptr, MROWS, 3 * INNER, DIM);
  k_rope_pack<<<49152, 256, 0, stream>>>(qkv, ct, st, qb, kb, vT);
  // 64 (b,h) * 16 blocks, 8 waves (qtiles) per block
  k_attention<<<1024, 256, 0, stream>>>(qb, kb, vT, xmask, ct, st, ao);
  // out = ao @ wout^T + b : M=8192 N=1024 K=1024 -> 128*16 = 2048 waves
  k_gemm_bf16<<<256, 256, 0, stream>>>(ao, wout_bf, out, bout, MROWS, DIM, INNER);
}
